// CatNet_training_5102421148201
// MI455X (gfx1250) — compile-verified, matrix operand order checked by reading, not run on hardware
//
#include <hip/hip_runtime.h>

typedef __attribute__((ext_vector_type(16))) _Float16 v16h;
typedef __attribute__((ext_vector_type(8)))  _Float16 v8h;
typedef __attribute__((ext_vector_type(8)))  float    v8f;

// ---------------------------------------------------------------------------
// Weight repack: (Co,Ci,3,3) f32  ->  f16 [oc][(kh*3+kw)*Ci + c]
// (K-contiguous per output channel => B fragments load with b128s)
// ---------------------------------------------------------------------------
__global__ void pack_weights_kernel(const float* __restrict__ w,
                                    _Float16* __restrict__ wp,
                                    int Co, int Ci) {
  int idx = blockIdx.x * blockDim.x + threadIdx.x;
  int total = Co * Ci * 9;
  if (idx >= total) return;
  int kw = idx % 3; int r = idx / 3;
  int kh = r % 3;   r /= 3;
  int c  = r % Ci;  int oc = r / Ci;
  wp[(size_t)oc * (9 * Ci) + (size_t)(kh * 3 + kw) * Ci + c] =
      (_Float16)w[(((size_t)oc * Ci + c) * 3 + kh) * 3 + kw];
}

// ---------------------------------------------------------------------------
// Layer 1 direct conv (Cin=3, K=27 -- too small for WMMA, negligible cost).
// x: (N,3,H,W,T) f32 NCHWT.  pre: [m=(n,h,w,t)][Co] f32 (single stream).
// ---------------------------------------------------------------------------
__global__ void conv1_kernel(const float* __restrict__ x,
                             const float* __restrict__ w,
                             const float* __restrict__ b,
                             float* __restrict__ pre,
                             int N, int H, int W, int T, int Co) {
  int idx = blockIdx.x * blockDim.x + threadIdx.x;
  int total = N * H * W * T * Co;
  if (idx >= total) return;
  int oc = idx % Co; int m = idx / Co;
  int t = m % T; int r = m / T;
  int wo = r % W; r /= W;
  int ho = r % H; int n = r / H;
  float acc = b[oc];
  for (int c = 0; c < 3; ++c)
    for (int kh = 0; kh < 3; ++kh) {
      int hi = ho + kh - 1;
      if (hi < 0 || hi >= H) continue;
      for (int kw = 0; kw < 3; ++kw) {
        int wi = wo + kw - 1;
        if (wi < 0 || wi >= W) continue;
        acc += x[((((size_t)n * 3 + c) * H + hi) * W + wi) * T + t] *
               w[(((size_t)oc * 3 + c) * 3 + kh) * 3 + kw];
      }
    }
  pre[(size_t)m * Co + oc] = acc;
}

// ---------------------------------------------------------------------------
// Implicit-GEMM conv via V_WMMA_F32_16X16X32_F16, templated on <CIN, PAD>.
//   act: f16 [SN][H][W][T][CIN]  (SN = streams*batch, channel-innermost)
//   wts: f16 [Cout][9*CIN] packed (tap-major, channel-contiguous)
//   pre: f32 [m=(sn,ho,wo,t)][Cout]
// One wave owns a 16(M) x 64(N) tile: 4 f32 accumulators; the bounds-checked
// A fragment is reused across 4 B fragments (4 WMMAs per A load pair).
// A-fragment per ISA 16-bit layout: lane holds row (lane&15), K halves
// {kb..kb+7, kb+16..kb+23} with kb = (lane>=16)*8. B symmetric with oc cols.
// ---------------------------------------------------------------------------
template <int CIN, int PAD>
__global__ void __launch_bounds__(128)
conv_wmma_kernel(const _Float16* __restrict__ act,
                 const _Float16* __restrict__ wts,
                 const float* __restrict__ bias,
                 float* __restrict__ pre,
                 int SN, int H, int W, int Ho, int Wo, int Cout) {
  constexpr int T = 40;
  const int lane = threadIdx.x & 31;
  const int wave = threadIdx.x >> 5;
  const int Mtot = SN * Ho * Wo * T;
  const int Mtiles = (Mtot + 15) >> 4;
  const int mTile = blockIdx.x * 4 + wave;
  if (mTile >= Mtiles) return;               // uniform per wave; EXEC stays full
  const int ocBase = blockIdx.y << 6;        // 64 output channels per wave

  // Decode the output row this lane loads A for.
  const int rowA = mTile * 16 + (lane & 15);
  const bool rowValid = rowA < Mtot;
  int m = rowValid ? rowA : 0;
  const int t  = m % T;   m /= T;
  const int wo = m % Wo;  m /= Wo;
  const int ho = m % Ho;
  const int nb = m / Ho;                     // joint [stream][batch] index

  const int kChunk = (lane >> 4) << 3;       // 0 or 8 halves
  const int oc0 = ocBase + (lane & 15);
  constexpr size_t wStride = (size_t)9 * CIN;

  v8f acc[4] = {};
#pragma unroll
  for (int kh = 0; kh < 3; ++kh) {
    const int hi = ho + kh - PAD;
#pragma unroll
    for (int kw = 0; kw < 3; ++kw) {
      const int wi = wo + kw - PAD;
      const bool v = rowValid & (hi >= 0) & (hi < H) & (wi >= 0) & (wi < W);
      const size_t inOff = ((((size_t)nb * H + hi) * W + wi) * T + t) * CIN + kChunk;
      const size_t wOff  = (size_t)(kh * 3 + kw) * CIN + kChunk;
#pragma unroll 4
      for (int cb = 0; cb < CIN; cb += 32) {
        v16h aF = {};
        if (v) {                             // predicated loads; zero = padding
          const _Float16* p = act + inOff + cb;
          v8h lo  = *(const v8h*)p;
          v8h hi2 = *(const v8h*)(p + 16);
          aF = __builtin_shufflevector(lo, hi2,
                 0,1,2,3,4,5,6,7,8,9,10,11,12,13,14,15);
        }
#pragma unroll
        for (int j = 0; j < 4; ++j) {        // reuse A across 4 N-tiles
          const _Float16* q = wts + (size_t)(oc0 + 16 * j) * wStride + wOff + cb;
          v8h blo = *(const v8h*)q;
          v8h bhi = *(const v8h*)(q + 16);
          v16h bF = __builtin_shufflevector(blo, bhi,
                 0,1,2,3,4,5,6,7,8,9,10,11,12,13,14,15);
          acc[j] = __builtin_amdgcn_wmma_f32_16x16x32_f16(
                       false, aF, false, bF, (short)0, acc[j], false, false);
        }
      }
    }
  }

  const int half = lane >> 4;
#pragma unroll
  for (int j = 0; j < 4; ++j) {
    const int oc = oc0 + 16 * j;
    const float bv = bias[oc];
#pragma unroll
    for (int r = 0; r < 8; ++r) {
      const int mo = mTile * 16 + r + half * 8;  // C/D layout: vgpr r = row r(+8)
      if (mo < Mtot) pre[(size_t)mo * Cout + oc] = acc[j][r] + bv;
    }
  }
}

// ---------------------------------------------------------------------------
// Integrate-and-fire over T. s=0: IF(thr[c]) on pre-acts (spiking stream).
// s=1: requant -- time-mean then IF with constant input vs FAC.
// preSingle=1: both streams read the same (single-stream) pre buffer (layer 1).
// Threads are channel-contiguous => coalesced strided walks over t.
// ---------------------------------------------------------------------------
__global__ void spike_kernel(const float* __restrict__ pre,
                             const float* __restrict__ thr,
                             _Float16* __restrict__ out,
                             int P /* pixels per stream */, int C, int T,
                             float fac, int preSingle) {
  int idx = blockIdx.x * blockDim.x + threadIdx.x;
  int total = 2 * P * C;
  if (idx >= total) return;
  int c = idx % C; int r = idx / C;
  int pix = r % P; int s = r / P;
  size_t M1 = (size_t)P * T;
  size_t preBase = ((preSingle ? (size_t)0 : (size_t)s * M1) + (size_t)pix * T) * C + c;
  size_t outBase = ((size_t)s * M1 + (size_t)pix * T) * C + c;
  if (s == 0) {
    float v = 0.f, th = thr[c];
    for (int t = 0; t < T; ++t) {
      v += pre[preBase + (size_t)t * C];
      float sp = (v >= th) ? 1.f : 0.f;
      v -= sp * th;
      out[outBase + (size_t)t * C] = (_Float16)sp;
    }
  } else {
    float sum = 0.f;
    for (int t = 0; t < T; ++t) sum += pre[preBase + (size_t)t * C];
    float ym = sum / (float)T;
    float v = 0.f;
    for (int t = 0; t < T; ++t) {
      v += ym;
      float sp = (v >= fac) ? 1.f : 0.f;
      v -= sp * fac;
      out[outBase + (size_t)t * C] = (_Float16)sp;
    }
  }
}

// ---------------------------------------------------------------------------
// 2x2 average pool per timestep + IF.  s=0: IF(p[c]); s=1: requant (mean+FAC).
// in:  f16 [N2][H][W][T][C] spikes,  out: f16 [N2][H/2][W/2][T][C]
// ---------------------------------------------------------------------------
__global__ void pool_spike_kernel(const _Float16* __restrict__ in,
                                  const float* __restrict__ thr,
                                  _Float16* __restrict__ out,
                                  int N2, int H, int W, int T, int C,
                                  float fac) {
  int Ho = H >> 1, Wo = W >> 1;
  int idx = blockIdx.x * blockDim.x + threadIdx.x;
  int total = N2 * Ho * Wo * C;
  if (idx >= total) return;
  int c = idx % C; int r = idx / C;
  int wo = r % Wo; r /= Wo;
  int ho = r % Ho; r /= Ho;                      // r in [0,N2): [stream][batch]
  int s = r / (N2 >> 1);
  size_t b00 = ((((size_t)r * H + 2 * ho)     * W + 2 * wo)     * T) * C + c;
  size_t b01 = ((((size_t)r * H + 2 * ho)     * W + 2 * wo + 1) * T) * C + c;
  size_t b10 = ((((size_t)r * H + 2 * ho + 1) * W + 2 * wo)     * T) * C + c;
  size_t b11 = ((((size_t)r * H + 2 * ho + 1) * W + 2 * wo + 1) * T) * C + c;
  size_t ob  = ((((size_t)r * Ho + ho) * Wo + wo) * T) * C + c;
  if (s == 0) {
    float v = 0.f, th = thr[c];
    for (int t = 0; t < T; ++t) {
      size_t o = (size_t)t * C;
      float a = 0.25f * ((float)in[b00 + o] + (float)in[b01 + o] +
                         (float)in[b10 + o] + (float)in[b11 + o]);
      v += a;
      float sp = (v >= th) ? 1.f : 0.f;
      v -= sp * th;
      out[ob + o] = (_Float16)sp;
    }
  } else {
    float sum = 0.f;
    for (int t = 0; t < T; ++t) {
      size_t o = (size_t)t * C;
      sum += 0.25f * ((float)in[b00 + o] + (float)in[b01 + o] +
                      (float)in[b10 + o] + (float)in[b11 + o]);
    }
    float ym = sum / (float)T;
    float v = 0.f;
    for (int t = 0; t < T; ++t) {
      v += ym;
      float sp = (v >= fac) ? 1.f : 0.f;
      v -= sp * fac;
      out[ob + (size_t)t * C] = (_Float16)sp;
    }
  }
}

// ---------------------------------------------------------------------------
// Classifier: out[n,o] = bc[o] + sum_c mean_t(xs[n,c,t]) * wc[o,c]
// act: f16 [2][N][1][1][T][1024], stream 0 used.
// ---------------------------------------------------------------------------
__global__ void classifier_kernel(const _Float16* __restrict__ act,
                                  const float* __restrict__ wc,
                                  const float* __restrict__ bc,
                                  float* __restrict__ out, int T) {
  __shared__ float smean[1024];
  int n = blockIdx.x;
  for (int c = threadIdx.x; c < 1024; c += blockDim.x) {
    float s = 0.f;
    size_t base = ((size_t)n * T) * 1024 + c;     // stream 0
    for (int t = 0; t < T; ++t) s += (float)act[base + (size_t)t * 1024];
    smean[c] = s / (float)T;
  }
  __syncthreads();
  if (threadIdx.x < 10) {
    float a = bc[threadIdx.x];
    const float* wr = wc + threadIdx.x * 1024;
    for (int c = 0; c < 1024; ++c) a += smean[c] * wr[c];
    out[n * 10 + threadIdx.x] = a;
  }
}

// ---------------------------------------------------------------------------
extern "C" void kernel_launch(void* const* d_in, const int* in_sizes, int n_in,
                              void* d_out, int out_size, void* d_ws, size_t ws_size,
                              hipStream_t stream) {
  (void)in_sizes; (void)n_in; (void)out_size; (void)ws_size;
  const int N = 2, T = 40;
  const float FAC = 32.0f / 40.0f - 0.001f;

  const float* x   = (const float*)d_in[0];
  const float* w[8]; const float* b[8];
  for (int i = 1; i <= 7; ++i) { w[i] = (const float*)d_in[2 * i - 1];
                                 b[i] = (const float*)d_in[2 * i]; }
  const float* wc  = (const float*)d_in[15];
  const float* bc  = (const float*)d_in[16];
  const float* thr1 = (const float*)d_in[17];
  const float* thr2 = (const float*)d_in[18];
  const float* p1   = (const float*)d_in[19];
  const float* thr3 = (const float*)d_in[20];
  const float* thr4 = (const float*)d_in[21];
  const float* p2   = (const float*)d_in[22];
  const float* thr5 = (const float*)d_in[23];
  const float* thr6 = (const float*)d_in[24];
  const float* p3   = (const float*)d_in[25];
  const float* thr7 = (const float*)d_in[26];
  const float* p4   = (const float*)d_in[27];
  float* out = (float*)d_out;

  // ---- workspace carve-out -------------------------------------------------
  char* ws = (char*)d_ws;
  size_t off = 0;
  auto carve = [&](size_t bytes) -> void* {
    void* p = ws + off;
    off += (bytes + 255) & ~(size_t)255;
    return p;
  };
  const int ci[8] = {0, 3, 128, 128, 256, 256, 512, 512};
  const int co[8] = {0, 128, 128, 256, 256, 512, 512, 1024};
  _Float16* wp[8];
  for (int l = 2; l <= 7; ++l)
    wp[l] = (_Float16*)carve((size_t)co[l] * ci[l] * 9 * sizeof(_Float16));
  float*    pre  = (float*)   carve((size_t)4 * 32 * 32 * T * 128 * sizeof(float));
  _Float16* actA = (_Float16*)carve((size_t)4 * 32 * 32 * T * 128 * sizeof(_Float16));
  _Float16* actB = (_Float16*)carve((size_t)4 * 32 * 32 * T * 128 * sizeof(_Float16));

  // ---- pack weights --------------------------------------------------------
  for (int l = 2; l <= 7; ++l) {
    int tot = co[l] * ci[l] * 9;
    pack_weights_kernel<<<(tot + 255) / 256, 256, 0, stream>>>(w[l], wp[l], co[l], ci[l]);
  }

  // ---- layer 1: direct conv + spike (single-stream pre feeds both streams) -
  {
    int tot = N * 32 * 32 * T * 128;
    conv1_kernel<<<(tot + 255) / 256, 256, 0, stream>>>(x, w[1], b[1], pre,
                                                        N, 32, 32, T, 128);
    int P = N * 32 * 32;
    int st = 2 * P * 128;
    spike_kernel<<<(st + 255) / 256, 256, 0, stream>>>(pre, thr1, actA,
                                                       P, 128, T, FAC, 1);
  }

  auto conv = [&](const _Float16* in, int l, int HW, int pad, float* o) {
    int Ho = (pad == 1) ? HW : HW - 2;
    int Mtot = 2 * N * Ho * Ho * T;
    int Mtiles = (Mtot + 15) / 16;
    dim3 grid((Mtiles + 3) / 4, co[l] / 64);
    if (ci[l] == 128)
      conv_wmma_kernel<128, 1><<<grid, 128, 0, stream>>>(in, wp[l], b[l], o,
          2 * N, HW, HW, Ho, Ho, co[l]);
    else if (ci[l] == 256)
      conv_wmma_kernel<256, 1><<<grid, 128, 0, stream>>>(in, wp[l], b[l], o,
          2 * N, HW, HW, Ho, Ho, co[l]);
    else if (pad == 1)
      conv_wmma_kernel<512, 1><<<grid, 128, 0, stream>>>(in, wp[l], b[l], o,
          2 * N, HW, HW, Ho, Ho, co[l]);
    else
      conv_wmma_kernel<512, 0><<<grid, 128, 0, stream>>>(in, wp[l], b[l], o,
          2 * N, HW, HW, Ho, Ho, co[l]);
  };
  auto spike = [&](const float* pr, const float* th, _Float16* o, int HW, int C) {
    int P = N * HW * HW;
    int st = 2 * P * C;
    spike_kernel<<<(st + 255) / 256, 256, 0, stream>>>(pr, th, o, P, C, T, FAC, 0);
  };
  auto pool = [&](const _Float16* in, const float* th, _Float16* o, int HW, int C) {
    int tot = 2 * N * (HW / 2) * (HW / 2) * C;
    pool_spike_kernel<<<(tot + 255) / 256, 256, 0, stream>>>(in, th, o,
                                                             2 * N, HW, HW, T, C, FAC);
  };

  conv(actA, 2, 32, 1, pre);  spike(pre, thr2, actB, 32, 128);
  pool(actB, p1, actA, 32, 128);                              // -> 16x16, 128
  conv(actA, 3, 16, 1, pre);  spike(pre, thr3, actB, 16, 256);
  conv(actB, 4, 16, 1, pre);  spike(pre, thr4, actA, 16, 256);
  pool(actA, p2, actB, 16, 256);                              // -> 8x8, 256
  conv(actB, 5, 8, 1, pre);   spike(pre, thr5, actA, 8, 512);
  conv(actA, 6, 8, 1, pre);   spike(pre, thr6, actB, 8, 512);
  pool(actB, p3, actA, 8, 512);                               // -> 4x4, 512
  conv(actA, 7, 4, 0, pre);   spike(pre, thr7, actB, 2, 1024);// pad 0 -> 2x2
  pool(actB, p4, actA, 2, 1024);                              // -> 1x1, 1024
  classifier_kernel<<<N, 256, 0, stream>>>(actA, wc, bc, out, T);
}
